// TransformerDecoderWithKVCache_67912022884443
// MI455X (gfx1250) — matrix operand order repaired and checked by reading, not compiled, not run on hardware
//
#include <hip/hip_runtime.h>
#include <hip/hip_bf16.h>
#include <math.h>

// ---------------------------------------------------------------------------
// Model constants (match reference)
// ---------------------------------------------------------------------------
#define NV  32000
#define ND  1024
#define NH  16
#define NL  4
#define NF  4096
#define NB  4
#define NS  1024
#define NDK 64
#define BSZ (NB*NS)          // 4096 rows of the activation stream

typedef __bf16 bf16_t;
typedef __attribute__((ext_vector_type(16))) __bf16 v16bf;
typedef __attribute__((ext_vector_type(8)))  __bf16 v8bf;
typedef __attribute__((ext_vector_type(8)))  float  v8f;
typedef __attribute__((ext_vector_type(4)))  int    v4i;

// ---------------------------------------------------------------------------
// gfx1250 async global->LDS copies (ASYNCcnt path), guarded so the build
// falls back to synchronous staging if the builtin is absent.
// Builtin signature (probe-confirmed via diagnostic):
//   void __builtin_amdgcn_global_load_async_to_lds_b128(AS1 v4i*, AS3 v4i*, Ii, Ii)
// ---------------------------------------------------------------------------
#define AS1 __attribute__((address_space(1)))
#define AS3 __attribute__((address_space(3)))

#if defined(__has_builtin)
#  if __has_builtin(__builtin_amdgcn_global_load_async_to_lds_b128)
#    define HAVE_ASYNC_LDS 1
#  endif
#endif

#ifdef HAVE_ASYNC_LDS
static __device__ __forceinline__ void async_cp16(const bf16_t* g, bf16_t* l)
{
    __builtin_amdgcn_global_load_async_to_lds_b128(
        (AS1 v4i*)(AS1 void*)(void*)const_cast<bf16_t*>(g),
        (AS3 v4i*)(AS3 void*)(void*)l, 0, 0);
}
static __device__ __forceinline__ void async_wait0()
{
    asm volatile("s_wait_asynccnt 0" ::: "memory");
}
#else
static __device__ __forceinline__ void async_cp16(const bf16_t* g, bf16_t* l)
{
    *(v8bf*)l = *(const v8bf*)g;
}
static __device__ __forceinline__ void async_wait0() {}
#endif

// ---------------------------------------------------------------------------
// WMMA fragment loaders (CDNA5 16-bit layouts, CDNA5 ISA §7.12.2)
//  A 16x32:  lane 0-15 row M=lane, K halves {0..7,16..23}; lane 16-31 {8..15,24..31}
//  B 32x16:  lane 0-15 col N=lane, K halves {0..7,16..23}; lane 16-31 {8..15,24..31} (B staged as [n][k])
// Both reduce to two contiguous 16B LDS loads per fragment.
// ---------------------------------------------------------------------------
static __device__ __forceinline__ v16bf frag_a(const bf16_t* base, int row0, int k0,
                                               int ld, int lane)
{
    int m  = row0 + (lane & 15);
    int kb = (lane < 16) ? 0 : 8;
    const bf16_t* p = base + m * ld + k0;
    v8bf lo = *(const v8bf*)(p + kb);
    v8bf hi = *(const v8bf*)(p + kb + 16);
    v16bf r;
#pragma unroll
    for (int e = 0; e < 8; ++e) { r[e] = lo[e]; r[8 + e] = hi[e]; }
    return r;
}

static __device__ __forceinline__ v16bf frag_b(const bf16_t* base, int n0, int k0,
                                               int ld, int lane)
{
    int n  = n0 + (lane & 15);
    int kb = (lane < 16) ? 0 : 8;
    const bf16_t* p = base + n * ld + k0 + kb;
    v8bf lo = *(const v8bf*)(p);
    v8bf hi = *(const v8bf*)(p + 16);
    v16bf r;
#pragma unroll
    for (int e = 0; e < 8; ++e) { r[e] = lo[e]; r[8 + e] = hi[e]; }
    return r;
}

// ---------------------------------------------------------------------------
// Generic bf16 WMMA GEMM: C[M,N] = A[M,K] * W[K,N] + bias, optional exact GELU,
// optional bf16 copy of C. Tiles: 128x128x64, 8 waves (4x2), 2x4 accums/wave.
// A tile staged via async global->LDS; W tile staged transposed via VGPRs.
// ---------------------------------------------------------------------------
#define TM 128
#define TN 128
#define TK 64

__global__ __launch_bounds__(256)
void gemm_bf16_wmma(const bf16_t* __restrict__ A, const bf16_t* __restrict__ W,
                    const float* __restrict__ bias, float* __restrict__ C,
                    bf16_t* __restrict__ Cb, int M, int N, int K, int fuseGelu)
{
    __shared__ __align__(16) bf16_t As[TM * TK];   // [m][k]
    __shared__ __align__(16) bf16_t Bt[TN * TK];   // [n][k]  (transposed stage)

    int tid  = threadIdx.x;
    int lane = tid & 31;
    int wid  = tid >> 5;
    int wm   = wid & 3;          // 4 waves along M (32 rows each)
    int wn   = wid >> 2;         // 2 waves along N (64 cols each)
    int row0 = blockIdx.y * TM;
    int col0 = blockIdx.x * TN;

    v8f acc[2][4];
#pragma unroll
    for (int i = 0; i < 2; ++i)
#pragma unroll
        for (int j = 0; j < 4; ++j)
#pragma unroll
            for (int r = 0; r < 8; ++r) acc[i][j][r] = 0.f;

    // Per-thread staging coordinates (256 threads move 128x64 A and 64x128 W)
    int ar = tid >> 1;               // A row 0..127
    int ac = (tid & 1) * 32;         // A col {0,32}
    int bk = tid >> 2;               // W k-row 0..63
    int bn = (tid & 3) * 32;         // W n-col {0,32,64,96}
    const bf16_t* ag = A + (size_t)(row0 + ar) * K + ac;
    const bf16_t* wg = W + (size_t)bk * N + col0 + bn;

    for (int k0 = 0; k0 < K; k0 += TK) {
        // A tile: 4 x 16B async copies straight into LDS (ASYNCcnt tracked)
#pragma unroll
        for (int e = 0; e < 4; ++e)
            async_cp16(ag + 8 * e, &As[ar * TK + ac + 8 * e]);

        // W tile: VGPR round-trip + transpose into [n][k]
        v8bf b0 = *(const v8bf*)(wg);
        v8bf b1 = *(const v8bf*)(wg + 8);
        v8bf b2 = *(const v8bf*)(wg + 16);
        v8bf b3 = *(const v8bf*)(wg + 24);
        ag += TK;
        wg += (size_t)TK * N;
        if (k0 + TK < K) {           // pull next tiles toward L2 (global_prefetch_b8)
            __builtin_prefetch(ag, 0, 1);
            __builtin_prefetch(wg, 0, 1);
        }
#pragma unroll
        for (int e = 0; e < 8; ++e) Bt[(bn + e)      * TK + bk] = b0[e];
#pragma unroll
        for (int e = 0; e < 8; ++e) Bt[(bn + 8 + e)  * TK + bk] = b1[e];
#pragma unroll
        for (int e = 0; e < 8; ++e) Bt[(bn + 16 + e) * TK + bk] = b2[e];
#pragma unroll
        for (int e = 0; e < 8; ++e) Bt[(bn + 24 + e) * TK + bk] = b3[e];

        async_wait0();
        __syncthreads();

#pragma unroll
        for (int kk = 0; kk < TK; kk += 32) {
            v16bf afr[2], bfr[4];
#pragma unroll
            for (int i = 0; i < 2; ++i)
                afr[i] = frag_a(As, wm * 32 + i * 16, kk, TK, lane);
#pragma unroll
            for (int j = 0; j < 4; ++j)
                bfr[j] = frag_b(Bt, wn * 64 + j * 16, kk, TK, lane);
#pragma unroll
            for (int i = 0; i < 2; ++i)
#pragma unroll
                for (int j = 0; j < 4; ++j)
                    acc[i][j] = __builtin_amdgcn_wmma_f32_16x16x32_bf16(
                        false, afr[i], false, bfr[j], (short)0, acc[i][j], false, false);
        }
        __syncthreads();
    }

    // Epilogue: C/D layout — lane(0..15)=N col, lane>>4 selects M half, vgpr r = row
    int mbase = (lane >> 4) * 8;
    int nlan  = lane & 15;
#pragma unroll
    for (int i = 0; i < 2; ++i)
#pragma unroll
        for (int j = 0; j < 4; ++j) {
            int   col = col0 + wn * 64 + j * 16 + nlan;
            float bv  = bias ? bias[col] : 0.f;
#pragma unroll
            for (int r = 0; r < 8; ++r) {
                int   row = row0 + wm * 32 + i * 16 + mbase + r;
                float v   = acc[i][j][r] + bv;
                if (fuseGelu) v = 0.5f * v * (1.f + erff(v * 0.70710678f));
                size_t idx = (size_t)row * N + col;
                C[idx] = v;
                if (Cb) Cb[idx] = (bf16_t)v;
            }
        }
}

// ---------------------------------------------------------------------------
// Flash-style causal attention. Q/K/V are bf16 [B,H,S,64]. One block = 64 query
// rows of one (b,h). Key tiles of 64, online softmax, both matmuls on WMMA.
// Q/K tiles staged via async global->LDS; V staged transposed via VGPRs.
// ---------------------------------------------------------------------------
__global__ __launch_bounds__(256)
void attn_kernel(const bf16_t* __restrict__ Q, const bf16_t* __restrict__ Kc,
                 const bf16_t* __restrict__ Vc, float* __restrict__ O)
{
    __shared__ __align__(16) bf16_t Qs[64 * 64];   // [q][d]
    __shared__ __align__(16) bf16_t Kt[64 * 64];   // [key][d] (== B-fragment layout)
    __shared__ __align__(16) bf16_t Vt[64 * 64];   // [d][key] (transposed stage)
    __shared__ __align__(16) bf16_t Ps[64 * 64];   // probs    [q][key]
    __shared__ __align__(16) float  Sc[64 * 64];   // scores   [q][key]
    __shared__ float mrow[64], lrow[64], fac[64];

    int tid  = threadIdx.x;
    int lane = tid & 31;
    int wid  = tid >> 5;
    int wm   = wid & 3;            // 4 waves over 64 q rows (16 each)
    int wn   = wid >> 2;           // 2 waves over key/d cols (32 each)
    int bh   = blockIdx.x;
    int q0   = blockIdx.y * 64;
    size_t base = (size_t)bh * NS * NDK;

    {   // stage Q tile 64x64 (async)
        int r = tid >> 2, c = (tid & 3) * 16;
        const bf16_t* g = Q + base + (size_t)(q0 + r) * NDK + c;
        async_cp16(g,     &Qs[r * 64 + c]);
        async_cp16(g + 8, &Qs[r * 64 + c + 8]);
    }
    if (tid < 64) { mrow[tid] = -1e30f; lrow[tid] = 0.f; }

    v8f oacc[2];
#pragma unroll
    for (int j = 0; j < 2; ++j)
#pragma unroll
        for (int r = 0; r < 8; ++r) oacc[j][r] = 0.f;
    async_wait0();
    __syncthreads();

    for (int kt = 0; kt < q0 + 64; kt += 64) {
        {   // stage K tile [key][d] (async)
            int r = tid >> 2, c = (tid & 3) * 16;
            const bf16_t* g = Kc + base + (size_t)(kt + r) * NDK + c;
            async_cp16(g,     &Kt[r * 64 + c]);
            async_cp16(g + 8, &Kt[r * 64 + c + 8]);
        }
        {   // stage V tile transposed -> [d][key]
            int r = tid >> 2, c = (tid & 3) * 16;
            const bf16_t* g = Vc + base + (size_t)(kt + r) * NDK + c;
            v8bf v0 = *(const v8bf*)(g);
            v8bf v1 = *(const v8bf*)(g + 8);
#pragma unroll
            for (int e = 0; e < 8; ++e) Vt[(c + e)     * 64 + r] = v0[e];
#pragma unroll
            for (int e = 0; e < 8; ++e) Vt[(c + 8 + e) * 64 + r] = v1[e];
        }
        async_wait0();
        __syncthreads();

        // scores: S = Q * K^T  (per wave: 16 q rows x 32 keys)
        v8f sc[2];
#pragma unroll
        for (int j = 0; j < 2; ++j)
#pragma unroll
            for (int r = 0; r < 8; ++r) sc[j][r] = 0.f;
#pragma unroll
        for (int d0 = 0; d0 < NDK; d0 += 32) {
            v16bf a = frag_a(Qs, 16 * wm, d0, 64, lane);
#pragma unroll
            for (int j = 0; j < 2; ++j) {
                v16bf b = frag_b(Kt, 32 * wn + 16 * j, d0, 64, lane);
                sc[j] = __builtin_amdgcn_wmma_f32_16x16x32_bf16(
                    false, a, false, b, (short)0, sc[j], false, false);
            }
        }
        {   // scale + causal mask -> Sc
            int mb = (lane >> 4) * 8, nl = lane & 15;
#pragma unroll
            for (int j = 0; j < 2; ++j) {
                int col = 32 * wn + 16 * j + nl;
                int kg  = kt + col;
#pragma unroll
                for (int r = 0; r < 8; ++r) {
                    int   row = 16 * wm + mb + r;
                    float s   = sc[j][r] * 0.125f;          // 1/sqrt(64)
                    if (kg > q0 + row) s = -1e30f;
                    Sc[row * 64 + col] = s;
                }
            }
        }
        __syncthreads();

        // online softmax (one thread per query row)
        if (tid < 64) {
            int   row  = tid;
            float mold = mrow[row];
            float tmax = -1e30f;
            for (int c = 0; c < 64; ++c) tmax = fmaxf(tmax, Sc[row * 64 + c]);
            float mnew  = fmaxf(mold, tmax);
            float alpha = __expf(mold - mnew);
            float sum   = 0.f;
            for (int c = 0; c < 64; ++c) {
                float p = __expf(Sc[row * 64 + c] - mnew);
                sum += p;
                Ps[row * 64 + c] = (bf16_t)p;
            }
            mrow[row] = mnew;
            lrow[row] = lrow[row] * alpha + sum;
            fac[row]  = alpha;
        }
        __syncthreads();

        // rescale running O, then O += P * V
        {
            int mb = (lane >> 4) * 8;
#pragma unroll
            for (int j = 0; j < 2; ++j)
#pragma unroll
                for (int r = 0; r < 8; ++r) oacc[j][r] *= fac[16 * wm + mb + r];
        }
#pragma unroll
        for (int ks = 0; ks < 2; ++ks) {
            v16bf a = frag_a(Ps, 16 * wm, 32 * ks, 64, lane);
#pragma unroll
            for (int j = 0; j < 2; ++j) {
                v16bf b = frag_b(Vt, 32 * wn + 16 * j, 32 * ks, 64, lane);
                oacc[j] = __builtin_amdgcn_wmma_f32_16x16x32_bf16(
                    false, a, false, b, (short)0, oacc[j], false, false);
            }
        }
        __syncthreads();
    }

    // normalize and write merged-head output [B,S,D] f32
    int b_ = bh / NH, h_ = bh % NH;
    int mb = (lane >> 4) * 8, nl = lane & 15;
#pragma unroll
    for (int j = 0; j < 2; ++j)
#pragma unroll
        for (int r = 0; r < 8; ++r) {
            int   row  = 16 * wm + mb + r;
            int   dcol = 32 * wn + 16 * j + nl;
            float v    = oacc[j][r] / lrow[row];
            O[((size_t)(b_ * NS + q0 + row)) * ND + h_ * NDK + dcol] = v;
        }
}

// ---------------------------------------------------------------------------
// Elementwise / helper kernels
// ---------------------------------------------------------------------------
__global__ void cvt_f32_bf16_kernel(const float* __restrict__ src,
                                    bf16_t* __restrict__ dst, size_t n)
{
    size_t i = (size_t)blockIdx.x * blockDim.x + threadIdx.x;
    if (i < n) dst[i] = (bf16_t)src[i];
}

__global__ void embed_kernel(const int* __restrict__ ids, const float* __restrict__ emb,
                             const float* __restrict__ pos, float* __restrict__ xF,
                             bf16_t* __restrict__ xB)
{
    size_t i = (size_t)blockIdx.x * blockDim.x + threadIdx.x;
    if (i >= (size_t)BSZ * ND) return;
    int    d  = (int)(i % ND);
    size_t bs = i / ND;
    int    s  = (int)(bs % NS);
    int    t  = ids[bs];
    float  v  = emb[(size_t)t * ND + d] + pos[(size_t)s * ND + d];
    xF[i] = v;
    xB[i] = (bf16_t)v;
}

__global__ void split_heads_kernel(const float* __restrict__ src,
                                   bf16_t* __restrict__ dst)
{
    size_t i = (size_t)blockIdx.x * blockDim.x + threadIdx.x;
    if (i >= (size_t)BSZ * ND) return;
    int    d  = (int)(i % ND);
    size_t bs = i / ND;
    int    b  = (int)(bs / NS), s = (int)(bs % NS);
    int    h  = d >> 6, dk = d & 63;
    dst[(((size_t)(b * NH + h) * NS + s) << 6) + dk] = (bf16_t)src[i];
}

__global__ __launch_bounds__(256)
void residual_ln_kernel(const float* __restrict__ X, const float* __restrict__ Yadd,
                        const float* __restrict__ g, const float* __restrict__ b,
                        float* __restrict__ outF, bf16_t* __restrict__ outB, int doAdd)
{
    __shared__ float red[256];
    int row = blockIdx.x, tid = threadIdx.x;
    const float* x = X    + (size_t)row * ND;
    const float* y = Yadd + (size_t)row * ND;

    float loc[ND / 256];
    float s = 0.f;
#pragma unroll
    for (int i = 0; i < ND / 256; ++i) {
        int   c = tid + i * 256;
        float v = x[c] + (doAdd ? y[c] : 0.f);
        loc[i] = v;
        s += v;
    }
    red[tid] = s;
    __syncthreads();
    for (int st = 128; st > 0; st >>= 1) {
        if (tid < st) red[tid] += red[tid + st];
        __syncthreads();
    }
    float mean = red[0] / ND;
    __syncthreads();

    float vs = 0.f;
#pragma unroll
    for (int i = 0; i < ND / 256; ++i) {
        float d = loc[i] - mean;
        vs += d * d;
    }
    red[tid] = vs;
    __syncthreads();
    for (int st = 128; st > 0; st >>= 1) {
        if (tid < st) red[tid] += red[tid + st];
        __syncthreads();
    }
    float rstd = rsqrtf(red[0] / ND + 1e-5f);

#pragma unroll
    for (int i = 0; i < ND / 256; ++i) {
        int   c = tid + i * 256;
        float n = (loc[i] - mean) * rstd * g[c] + b[c];
        outF[(size_t)row * ND + c] = n;
        outB[(size_t)row * ND + c] = (bf16_t)n;
    }
}

// ---------------------------------------------------------------------------
// Host orchestration
// ---------------------------------------------------------------------------
extern "C" void kernel_launch(void* const* d_in, const int* in_sizes, int n_in,
                              void* d_out, int out_size, void* d_ws, size_t ws_size,
                              hipStream_t stream)
{
    (void)in_sizes; (void)n_in; (void)out_size; (void)ws_size;

    const int*   ids   = (const int*)  d_in[0];
    const float* emb   = (const float*)d_in[1];
    const float* pos   = (const float*)d_in[2];
    const float* wq    = (const float*)d_in[3];
    const float* bq    = (const float*)d_in[4];
    const float* wk    = (const float*)d_in[5];
    const float* bk    = (const float*)d_in[6];
    const float* wv    = (const float*)d_in[7];
    const float* bv    = (const float*)d_in[8];
    const float* wo    = (const float*)d_in[9];
    const float* bo    = (const float*)d_in[10];
    const float* ln1_g = (const float*)d_in[11];
    const float* ln1_b = (const float*)d_in[12];
    const float* ln2_g = (const float*)d_in[13];
    const float* ln2_b = (const float*)d_in[14];
    const float* w1    = (const float*)d_in[15];
    const float* b1    = (const float*)d_in[16];
    const float* w2    = (const float*)d_in[17];
    const float* b2    = (const float*)d_in[18];
    const float* lnf_g = (const float*)d_in[19];
    const float* lnf_b = (const float*)d_in[20];
    const float* w_out = (const float*)d_in[21];
    const float* b_out = (const float*)d_in[22];
    float* out = (float*)d_out;

    const size_t DD  = (size_t)ND * ND;
    const size_t DF  = (size_t)ND * NF;
    const size_t DV  = (size_t)ND * NV;
    const size_t BSD = (size_t)BSZ * ND;
    const size_t BSF = (size_t)BSZ * NF;

    char*  ws  = (char*)d_ws;
    size_t off = 0;
    auto take = [&](size_t bytes) -> char* {
        char* p = ws + off;
        off += (bytes + 255) & ~(size_t)255;
        return p;
    };

    bf16_t* wqB   = (bf16_t*)take(NL * DD * 2);
    bf16_t* wkB   = (bf16_t*)take(NL * DD * 2);
    bf16_t* wvB   = (bf16_t*)take(NL * DD * 2);
    bf16_t* woB   = (bf16_t*)take(NL * DD * 2);
    bf16_t* w1B   = (bf16_t*)take(NL * DF * 2);
    bf16_t* w2B   = (bf16_t*)take(NL * DF * 2);
    bf16_t* woutB = (bf16_t*)take(DV * 2);
    float*  xF    = (float*) take(BSD * 4);
    bf16_t* xB    = (bf16_t*)take(BSD * 2);
    bf16_t* qB    = (bf16_t*)take(BSD * 2);
    bf16_t* kB    = (bf16_t*)take(BSD * 2);
    bf16_t* vB    = (bf16_t*)take(BSD * 2);
    float*  aoF   = (float*) take(BSD * 4);
    bf16_t* aoB   = (bf16_t*)take(BSD * 2);
    float*  tF    = (float*) take(BSF * 4);   // shared f32 GEMM output scratch
    bf16_t* hB    = (bf16_t*)take(BSF * 2);   // FFN hidden (bf16)

    dim3 blk(256);
    auto cvt = [&](const float* s, bf16_t* d, size_t n) {
        cvt_f32_bf16_kernel<<<dim3((unsigned)((n + 255) / 256)), blk, 0, stream>>>(s, d, n);
    };

    // One-time (per call) weight downconversion f32 -> bf16
    cvt(wq, wqB, NL * DD);
    cvt(wk, wkB, NL * DD);
    cvt(wv, wvB, NL * DD);
    cvt(wo, woB, NL * DD);
    cvt(w1, w1B, NL * DF);
    cvt(w2, w2B, NL * DF);
    cvt(w_out, woutB, DV);

    embed_kernel<<<dim3((unsigned)((BSD + 255) / 256)), blk, 0, stream>>>(ids, emb, pos, xF, xB);

    dim3 gD((unsigned)(ND / TN), (unsigned)(BSZ / TM));   // N=1024 GEMMs
    dim3 gF((unsigned)(NF / TN), (unsigned)(BSZ / TM));   // N=4096 GEMM
    dim3 gE((unsigned)((BSD + 255) / 256));

    for (int l = 0; l < NL; ++l) {
        // Q, K, V projections -> split heads to bf16 [B,H,S,64]
        gemm_bf16_wmma<<<gD, blk, 0, stream>>>(xB, wqB + (size_t)l * DD, bq + (size_t)l * ND,
                                               tF, nullptr, BSZ, ND, ND, 0);
        split_heads_kernel<<<gE, blk, 0, stream>>>(tF, qB);
        gemm_bf16_wmma<<<gD, blk, 0, stream>>>(xB, wkB + (size_t)l * DD, bk + (size_t)l * ND,
                                               tF, nullptr, BSZ, ND, ND, 0);
        split_heads_kernel<<<gE, blk, 0, stream>>>(tF, kB);
        gemm_bf16_wmma<<<gD, blk, 0, stream>>>(xB, wvB + (size_t)l * DD, bv + (size_t)l * ND,
                                               tF, nullptr, BSZ, ND, ND, 0);
        split_heads_kernel<<<gE, blk, 0, stream>>>(tF, vB);

        // Causal attention (flash-style, WMMA both matmuls)
        attn_kernel<<<dim3(NB * NH, NS / 64), blk, 0, stream>>>(qB, kB, vB, aoF);
        cvt(aoF, aoB, BSD);

        // Output projection, residual + LN1
        gemm_bf16_wmma<<<gD, blk, 0, stream>>>(aoB, woB + (size_t)l * DD, bo + (size_t)l * ND,
                                               tF, nullptr, BSZ, ND, ND, 0);
        residual_ln_kernel<<<dim3((unsigned)BSZ), blk, 0, stream>>>(
            xF, tF, ln1_g + (size_t)l * ND, ln1_b + (size_t)l * ND, xF, xB, 1);

        // FFN: GELU fused into first GEMM epilogue (also emits bf16 hidden)
        gemm_bf16_wmma<<<gF, blk, 0, stream>>>(xB, w1B + (size_t)l * DF, b1 + (size_t)l * NF,
                                               tF, hB, BSZ, NF, ND, 1);
        gemm_bf16_wmma<<<gD, blk, 0, stream>>>(hB, w2B + (size_t)l * DF, b2 + (size_t)l * ND,
                                               tF, nullptr, BSZ, ND, NF, 0);
        residual_ln_kernel<<<dim3((unsigned)BSZ), blk, 0, stream>>>(
            xF, tF, ln2_g + (size_t)l * ND, ln2_b + (size_t)l * ND, xF, xB, 1);
    }

    // Final LN, then vocab projection straight into d_out (f32)
    residual_ln_kernel<<<dim3((unsigned)BSZ), blk, 0, stream>>>(
        xF, xF, lnf_g, lnf_b, xF, xB, 0);
    dim3 gV((unsigned)(NV / TN), (unsigned)(BSZ / TM));
    gemm_bf16_wmma<<<gV, blk, 0, stream>>>(xB, woutB, b_out, out, nullptr, BSZ, NV, ND, 0);
}
